// EdgeNetwork_69415261438419
// MI455X (gfx1250) — matrix operand order repaired and checked by reading, not compiled, not run on hardware
//
#include <hip/hip_runtime.h>
#include <hip/hip_bf16.h>

// ---------------------------------------------------------------------------
// Fused edge-gather + 3-layer MLP for MI455X (gfx1250), bf16 WMMA path.
//   x[E,321] = concat(dst_feat, dst_hid, src_feat, dst_hid, dist)
//   out = relu(relu(x@W1+b1)@W2+b2)@W3+b3
// 32-edge tiles staged in LDS as bf16; GEMMs via v_wmma_f32_16x16x32_bf16
// (fp32 accum); weights pre-packed into WMMA B-fragment order in workspace.
// K-loops fully unrolled so the scheduler can pipeline L2 fragment loads
// across WMMA bursts; streaming data uses non-temporal hints to keep the
// reused node tables + weights resident in the 192MB L2.
// ---------------------------------------------------------------------------

typedef __attribute__((ext_vector_type(16))) __bf16 v16bf;
typedef __attribute__((ext_vector_type(8)))  __bf16 v8bf;
typedef __attribute__((ext_vector_type(8)))  float  v8f;

#define MB      32    // edges (rows) per workgroup
#define STRIDE  512   // LDS row stride in bf16 elements
#define INP     352   // IN_DIM 321 padded to 11*32
#define OUTD    256
#define KT1     11    // K tiles (K=32 each) for layer 1
#define KT2     16
#define KT3     16
#define NTT1    32    // total N tiles (N=16 each)
#define NTT2    32
#define NTT3    16

struct bfpair { v8bf lo, hi; };

// XOR-swizzle 16B granules within a row so that the 16 rows of an A-fragment
// hit distinct bank groups (row stride is 1024B == bank-period multiple).
__device__ __forceinline__ int swz(int row, int col) {
  return row * STRIDE + ((((col >> 3) ^ (row & 15)) << 3) | (col & 7));
}

// A-fragment (16x32 bf16) per ISA layout:
//   lanes 0-15 : M=lane,    K = kbase+{0..7} then kbase+{16..23}
//   lanes 16-31: M=lane-16, K = kbase+{8..15} then kbase+{24..31}
__device__ __forceinline__ v16bf load_a(const __bf16* buf, int rbase, int lane, int kbase) {
  int row = rbase + (lane & 15);
  int ko  = kbase + ((lane >> 4) << 3);
  bfpair p;
  p.lo = *(const v8bf*)(buf + swz(row, ko));
  p.hi = *(const v8bf*)(buf + swz(row, ko + 16));
  return __builtin_bit_cast(v16bf, p);
}

// B-fragment pre-packed: 32 lanes x 16 bf16 contiguous per (ktile,ntile).
__device__ __forceinline__ v16bf load_b(const __bf16* frag, int lane) {
  const v8bf* p8 = (const v8bf*)(frag + lane * 16);
  bfpair p;
  p.lo = p8[0];
  p.hi = p8[1];
  return __builtin_bit_cast(v16bf, p);
}

// One GEMM layer for one wave: rows [rbase, rbase+16), N tiles
// [ntBase, ntBase+NTW). C/D f32 layout: VGPR v, lanes 0-15 -> (M=v, N=lane),
// lanes 16-31 -> (M=8+v, N=lane-16).
template<int KT, int NTW, bool RELU, bool TO_LDS>
__device__ __forceinline__ void gemm_layer(
    const __bf16* inbuf, const __bf16* __restrict__ pW, const float* __restrict__ bias,
    int NTtot, int ntBase, int rbase, int lane,
    __bf16* outbuf, float* __restrict__ gout, long e0, int E)
{
  v8f acc[NTW];
#pragma unroll
  for (int nt = 0; nt < NTW; ++nt) {
    float bv = bias[(ntBase + nt) * 16 + (lane & 15)];
    v8f c;
#pragma unroll
    for (int i = 0; i < 8; ++i) c[i] = bv;
    acc[nt] = c;
  }

  // Fully unrolled K loop: batch the fragment loads, then a WMMA burst, so
  // the scheduler can overlap kt+1 loads with kt matrix ops.
#pragma unroll
  for (int kt = 0; kt < KT; ++kt) {
    v16bf a = load_a(inbuf, rbase, lane, kt * 32);
    v16bf b[NTW];
#pragma unroll
    for (int nt = 0; nt < NTW; ++nt)
      b[nt] = load_b(pW + ((size_t)(kt * NTtot + ntBase + nt) << 9), lane);
#pragma unroll
    for (int nt = 0; nt < NTW; ++nt)
      acc[nt] = __builtin_amdgcn_wmma_f32_16x16x32_bf16(
          /*neg_a=*/false, a, /*neg_b=*/false, b[nt],
          /*c_mod=*/(short)0, acc[nt], /*reuse_a=*/false, /*reuse_b=*/false);
  }

  int mhi = (lane >> 4) << 3;   // +8 rows for high half-wave
  int n   = lane & 15;
#pragma unroll
  for (int nt = 0; nt < NTW; ++nt) {
    int col = (ntBase + nt) * 16 + n;
#pragma unroll
    for (int v = 0; v < 8; ++v) {
      float f = acc[nt][v];
      if (RELU) f = fmaxf(f, 0.0f);
      int row = rbase + mhi + v;
      if (TO_LDS) {
        outbuf[swz(row, col)] = (__bf16)f;
      } else {
        long e = e0 + row;
        if (e < E) {
          // Output is written once and never re-read: stream past L2.
          __builtin_nontemporal_store(f, &gout[e * OUTD + col]);
        }
      }
    }
  }
}

__global__ __launch_bounds__(256) void edge_mlp_kernel(
    const float* __restrict__ nf, const float* __restrict__ nh,
    const int* __restrict__ srcI, const int* __restrict__ dstI,
    const float* __restrict__ dist,
    const __bf16* __restrict__ pW1, const float* __restrict__ b1,
    const __bf16* __restrict__ pW2, const float* __restrict__ b2,
    const __bf16* __restrict__ pW3, const float* __restrict__ b3,
    float* __restrict__ out, int E)
{
  __shared__ __align__(16) __bf16 buf0[MB * STRIDE];  // 32 KB
  __shared__ __align__(16) __bf16 buf1[MB * STRIDE];  // 32 KB

  int tid  = threadIdx.x;
  int lane = tid & 31;
  int wave = tid >> 5;
  long e0  = (long)blockIdx.x * MB;

  // ---- Phase 1: gather x -> buf0 (bf16), cols 321..351 zero-padded.
  for (int i = tid; i < MB * INP; i += 256) {
    int r = i / INP;
    int c = i - r * INP;
    long e = e0 + r;
    float v = 0.0f;
    if (e < E) {
      int d = __builtin_nontemporal_load(&dstI[e]);
      if (c < 32)        v = nf[(long)d * 32 + c];
      else if (c < 160)  v = nh[(long)d * 128 + (c - 32)];
      else if (c < 192)  v = nf[(long)__builtin_nontemporal_load(&srcI[e]) * 32 + (c - 160)];
      else if (c < 320)  v = nh[(long)d * 128 + (c - 192)];
      else if (c == 320) v = __builtin_nontemporal_load(&dist[e]);
    }
    buf0[swz(r, c)] = (__bf16)v;
  }
  __syncthreads();

  int slab  = wave >> 2;   // 0..1 -> 16-row slab
  int nq    = wave & 3;    // 0..3 -> N quarter
  int rbase = slab * 16;

  // Layer 1: [32,352] x [352,512] -> buf1 (relu, bf16)
  gemm_layer<KT1, 8, true, true>(buf0, pW1, b1, NTT1, nq * 8, rbase, lane,
                                 buf1, nullptr, e0, E);
  __syncthreads();
  // Layer 2: [32,512] x [512,512] -> buf0 (relu, bf16)
  gemm_layer<KT2, 8, true, true>(buf1, pW2, b2, NTT2, nq * 8, rbase, lane,
                                 buf0, nullptr, e0, E);
  __syncthreads();
  // Layer 3: [32,512] x [512,256] -> out (fp32, global)
  gemm_layer<KT3, 4, false, false>(buf0, pW3, b3, NTT3, nq * 4, rbase, lane,
                                   nullptr, out, e0, E);
}

// ---------------------------------------------------------------------------
// Repack fp32 row-major W[K,N] -> bf16 WMMA B-fragments.
// Fragment (kt,nt): lane l holds column n = nt*16 + (l&15),
//   K = kt*32 + (l>=16 ? 8 : 0) + s + (s>=8 ? 8 : 0), s = 0..15, stored contiguously.
// One thread per (kt,nt,lane); zero-fills K/N padding.
// ---------------------------------------------------------------------------
__global__ void repack_kernel(const float* __restrict__ W, __bf16* __restrict__ dst,
                              int K, int N, int KT, int NT)
{
  int idx = blockIdx.x * 256 + threadIdx.x;
  int total = KT * NT * 32;
  if (idx >= total) return;
  int lane = idx & 31;
  int fi   = idx >> 5;
  int nt   = fi % NT;
  int kt   = fi / NT;
  int n  = nt * 16 + (lane & 15);
  int kb = kt * 32 + ((lane >> 4) << 3);
  __bf16* o = dst + (size_t)idx * 16;
#pragma unroll
  for (int s = 0; s < 16; ++s) {
    int k = kb + s + ((s >> 3) << 3);
    float v = (k < K && n < N) ? W[(size_t)k * N + n] : 0.0f;
    o[s] = (__bf16)v;
  }
}

extern "C" void kernel_launch(void* const* d_in, const int* in_sizes, int n_in,
                              void* d_out, int out_size, void* d_ws, size_t ws_size,
                              hipStream_t stream) {
  const float* nf   = (const float*)d_in[0];
  const float* nh   = (const float*)d_in[1];
  const int*   srcI = (const int*)d_in[2];
  const int*   dstI = (const int*)d_in[3];
  const float* dist = (const float*)d_in[4];
  const float* W1   = (const float*)d_in[5];
  const float* b1   = (const float*)d_in[6];
  const float* W2   = (const float*)d_in[7];
  const float* b2   = (const float*)d_in[8];
  const float* W3   = (const float*)d_in[9];
  const float* b3   = (const float*)d_in[10];
  float* out = (float*)d_out;
  int E = in_sizes[2];

  // Workspace: packed bf16 weights (total ~1.12 MB).
  char* ws = (char*)d_ws;
  __bf16* pW1 = (__bf16*)(ws);                       // 11*32*512*2 = 360448 B
  __bf16* pW2 = (__bf16*)(ws + 360448);              // 16*32*512*2 = 524288 B
  __bf16* pW3 = (__bf16*)(ws + 360448 + 524288);     // 16*16*512*2 = 262144 B

  repack_kernel<<<(KT1 * NTT1 * 32 + 255) / 256, 256, 0, stream>>>(W1, pW1, 321, 512, KT1, NTT1);
  repack_kernel<<<(KT2 * NTT2 * 32 + 255) / 256, 256, 0, stream>>>(W2, pW2, 512, 512, KT2, NTT2);
  repack_kernel<<<(KT3 * NTT3 * 32 + 255) / 256, 256, 0, stream>>>(W3, pW3, 512, 256, KT3, NTT3);

  int blocks = (E + MB - 1) / MB;
  edge_mlp_kernel<<<blocks, 256, 0, stream>>>(nf, nh, srcI, dstI, dist,
                                              pW1, b1, pW2, b2, pW3, b3, out, E);
}
